// WeightedSupConLoss_61263413510463
// MI455X (gfx1250) — matrix operand order repaired and checked by reading, not compiled, not run on hardware
//
#include <hip/hip_runtime.h>

// ---------------------------------------------------------------------------
// Weighted SupCon loss, fused flash-style on gfx1250 (wave32, WMMA bf16).
//   B=4096 rows, D=1024 features, C=16 classes, T=baseT=0.1 -> T/baseT = 1.
// Rows are L2-normalized so sim = <f_i,f_j>/T in [-10,10]; the softmax shift
// cancels analytically, so a FIXED shift of 10 replaces the streaming max:
//   S = sum exp(sim-10), W = sum w, P = sum w*sim,  w = simw[i, labels[j]]
//   mlpp = (P - W*(10 + log(S+eps))) / (W+eps),  loss = mean_i(-mlpp)
// v3: 2x2 register tile (32-row stripe, 2 column tiles per wave) so each
// A-fragment (LDS) and each B-fragment (L2) feeds two WMMAs -> both the LDS
// and the L2 feed per v_wmma are halved vs the 1x1 tiling.
// ---------------------------------------------------------------------------

typedef __bf16 bf16_t;
typedef __attribute__((ext_vector_type(16))) __bf16 v16bf;
typedef __attribute__((ext_vector_type(8)))  __bf16 v8bf;
typedef __attribute__((ext_vector_type(4)))  __bf16 v4bf;
typedef __attribute__((ext_vector_type(8)))  float  v8f;
typedef __attribute__((ext_vector_type(4)))  float  v4f;

#define N_B    4096
#define K_D    1024
#define N_CLS  16
#define T_INV  10.0f      // 1 / TEMPERATURE
#define SHIFT  10.0f      // fixed softmax shift (upper bound of sim)
#define EPSF   1e-12f

// Dynamic-LDS layout (total 70656 B; WGP supports up to 320 KB/workgroup)
#define LDS_A_BYTES   65536            // 32 rows x 1024 bf16
#define LDS_SW_OFF    65536            // float swrowT[16][32] = 2 KB
#define LDS_STS_OFF   67584            // float st_s[8][32]    = 1 KB
#define LDS_STW_OFF   68608            // float st_w[8][32]    = 1 KB
#define LDS_STP_OFF   69632            // float st_p[8][32]    = 1 KB
#define LDS_TOTAL     70656

// ---------------------------------------------------------------------------
__global__ void wsc_zero_out(float* out) {
  if (threadIdx.x == 0 && blockIdx.x == 0) out[0] = 0.0f;
}

// ---------------------------------------------------------------------------
// Kernel 1: L2-normalize each row (f32 in) -> bf16 out (workspace).
__global__ __launch_bounds__(256)
void wsc_normalize(const float* __restrict__ feat, bf16_t* __restrict__ out) {
  const int row = blockIdx.x;
  const int tid = threadIdx.x;
  const float4 v = ((const float4*)(feat + (size_t)row * K_D))[tid];
  float ss = v.x * v.x + v.y * v.y + v.z * v.z + v.w * v.w;
#pragma unroll
  for (int off = 16; off > 0; off >>= 1) ss += __shfl_xor(ss, off, 32);
  __shared__ float red[8];
  if ((tid & 31) == 0) red[tid >> 5] = ss;
  __syncthreads();
  if (tid == 0) {
    float t = 0.0f;
#pragma unroll
    for (int i = 0; i < 8; ++i) t += red[i];
    red[0] = 1.0f / fmaxf(sqrtf(t), 1e-12f);   // F.normalize eps clamp
  }
  __syncthreads();
  const float s = red[0];
  v4bf o;
  o[0] = (bf16_t)(v.x * s); o[1] = (bf16_t)(v.y * s);
  o[2] = (bf16_t)(v.z * s); o[3] = (bf16_t)(v.w * s);
  ((v4bf*)(out + (size_t)row * K_D))[tid] = o;
}

// ---------------------------------------------------------------------------
// Fragment loaders (wave32 WMMA bf16 layouts).
__device__ __forceinline__ v16bf load_fragA(const bf16_t* row, int kk, int hi) {
  // A 16x32: lane halves hold K {0..7,16..23} / {8..15,24..31}
  const v8bf a0 = *(const v8bf*)(row + kk + hi * 8);
  const v8bf a1 = *(const v8bf*)(row + kk + 16 + hi * 8);
  v16bf A;
#pragma unroll
  for (int i = 0; i < 8; ++i) { A[i] = a0[i]; A[8 + i] = a1[i]; }
  return A;
}
__device__ __forceinline__ v16bf load_fragB(const bf16_t* row, int kk, int hi) {
  // B 32x16: lane = column; lane halves hold K {0..7,16..23} / {8..15,24..31}
  const v8bf b0 = *(const v8bf*)(row + kk + hi * 8);
  const v8bf b1 = *(const v8bf*)(row + kk + 16 + hi * 8);
  v16bf B;
#pragma unroll
  for (int i = 0; i < 8; ++i) { B[i] = b0[i]; B[8 + i] = b1[i]; }
  return B;
}

// Streaming-stat update for one 16x16 tile (diag_tile is wave-uniform).
__device__ __forceinline__ void tile_update(const v8f& acc, const float* wv,
                                            bool diag_tile, int m16, int hi,
                                            float (&s)[8], float (&w)[8],
                                            float (&p)[8]) {
  if (!diag_tile) {
#pragma unroll
    for (int k = 0; k < 8; ++k) {
      const float x = acc[k] * T_INV;
      const float e = __expf(x - SHIFT);
      s[k] += e;
      w[k] += wv[k];
      p[k] = fmaf(wv[k], x, p[k]);
    }
  } else {
#pragma unroll
    for (int k = 0; k < 8; ++k) {
      const bool d   = (hi * 8 + k) == m16;
      const float x  = acc[k] * T_INV;
      const float e  = d ? 0.0f : __expf(x - SHIFT);
      const float wk = d ? 0.0f : wv[k];
      s[k] += e;
      w[k] += wk;
      p[k] = fmaf(wk, x, p[k]);
    }
  }
}

// ---------------------------------------------------------------------------
// Kernel 2: fused sim-GEMM + streaming stats + loss reduction.
// Grid: 128 blocks (one 32-row stripe), 256 threads = 8 waves.
// Wave w sweeps column-tile pairs t = it*16 + 2w (+1), it = 0..15.
__global__ __launch_bounds__(256)
void wsc_main(const bf16_t* __restrict__ fbf, const int* __restrict__ labels,
              const float* __restrict__ simw, float* __restrict__ out) {
  extern __shared__ __align__(16) char smem[];
  v8bf*  As8    = (v8bf*)smem;                         // 32 x 1024 bf16
  float (*swrowT)[32] = (float(*)[32])(smem + LDS_SW_OFF);   // [class][row]
  float (*st_s)[32]   = (float(*)[32])(smem + LDS_STS_OFF);
  float (*st_w)[32]   = (float(*)[32])(smem + LDS_STW_OFF);
  float (*st_p)[32]   = (float(*)[32])(smem + LDS_STP_OFF);

  const int tid  = threadIdx.x;
  const int wave = tid >> 5;
  const int lane = tid & 31;
  const int m16  = lane & 15;
  const int hi   = lane >> 4;                  // 0: lanes 0-15, 1: lanes 16-31
  const int row_base = blockIdx.x * 32;

  // Stage A stripe into LDS: 4096 x 16B chunks, 16 per thread.
  {
    const v8bf* src = (const v8bf*)(fbf + (size_t)row_base * K_D);
#pragma unroll
    for (int i = 0; i < 16; ++i) As8[tid + 256 * i] = src[tid + 256 * i];
  }
  // Transposed weight table: swrowT[c][r] = simw[row_base+r][c] (512 entries).
#pragma unroll
  for (int i = 0; i < 2; ++i) {
    const int idx = tid + 256 * i;
    swrowT[idx >> 5][idx & 31] =
        simw[(size_t)(row_base + (idx & 31)) * N_CLS + (idx >> 5)];
  }
  __syncthreads();

  const bf16_t* aRow0 = (const bf16_t*)As8 + m16 * K_D;          // rows 0..15
  const bf16_t* aRow1 = (const bf16_t*)As8 + (m16 + 16) * K_D;   // rows 16..31

  // Stats per row-subtile h and VGPR slot k: global row = row_base+16h+8hi+k.
  float sk[2][8], wk[2][8], pk[2][8];
#pragma unroll
  for (int h = 0; h < 2; ++h)
#pragma unroll
    for (int k = 0; k < 8; ++k) { sk[h][k] = 0.0f; wk[h][k] = 0.0f; pk[h][k] = 0.0f; }

  for (int it = 0; it < 16; ++it) {
    const int cblk0 = (it * 16 + wave * 2) * 16;
    const int cblk1 = cblk0 + 16;
    const int col0  = cblk0 + m16;
    const int col1  = cblk1 + m16;
    const bf16_t* bRow0 = fbf + (size_t)col0 * K_D;
    const bf16_t* bRow1 = fbf + (size_t)col1 * K_D;

    v8f acc00 = {}, acc01 = {}, acc10 = {}, acc11 = {};
#pragma unroll 2
    for (int kk = 0; kk < K_D; kk += 32) {
      const v16bf A0 = load_fragA(aRow0, kk, hi);
      const v16bf A1 = load_fragA(aRow1, kk, hi);
      const v16bf B0 = load_fragB(bRow0, kk, hi);
      const v16bf B1 = load_fragB(bRow1, kk, hi);
      acc00 = __builtin_amdgcn_wmma_f32_16x16x32_bf16(false, A0, false, B0,
                                                      (short)0, acc00, false, false);
      acc01 = __builtin_amdgcn_wmma_f32_16x16x32_bf16(false, A0, false, B1,
                                                      (short)0, acc01, false, false);
      acc10 = __builtin_amdgcn_wmma_f32_16x16x32_bf16(false, A1, false, B0,
                                                      (short)0, acc10, false, false);
      acc11 = __builtin_amdgcn_wmma_f32_16x16x32_bf16(false, A1, false, B1,
                                                      (short)0, acc11, false, false);
    }

    // Weight vectors (loaded after the GEMM so they don't live across it):
    // wv[c][h][k] = simw[row_base+16h+8hi+k][labels[col_c]]
    const int lab0 = labels[col0] & (N_CLS - 1);
    const int lab1 = labels[col1] & (N_CLS - 1);
    float wv[2][2][8];
#pragma unroll
    for (int c = 0; c < 2; ++c) {
      const int lab = c ? lab1 : lab0;
#pragma unroll
      for (int h = 0; h < 2; ++h) {
        const v4f* wp = (const v4f*)&swrowT[lab][h * 16 + hi * 8];
        const v4f lo = wp[0], hi4 = wp[1];
#pragma unroll
        for (int i = 0; i < 4; ++i) { wv[c][h][i] = lo[i]; wv[c][h][4 + i] = hi4[i]; }
      }
    }

    // Diagonal can only appear when the column tile matches the row sub-tile.
    tile_update(acc00, wv[0][0], cblk0 == row_base,      m16, hi, sk[0], wk[0], pk[0]);
    tile_update(acc01, wv[1][0], cblk1 == row_base,      m16, hi, sk[0], wk[0], pk[0]);
    tile_update(acc10, wv[0][1], cblk0 == row_base + 16, m16, hi, sk[1], wk[1], pk[1]);
    tile_update(acc11, wv[1][1], cblk1 == row_base + 16, m16, hi, sk[1], wk[1], pk[1]);
  }

  // Reduce each row slot across the 16 lanes of each half (sums only).
#pragma unroll
  for (int h = 0; h < 2; ++h)
#pragma unroll
    for (int k = 0; k < 8; ++k) {
      float s = sk[h][k], w = wk[h][k], p = pk[h][k];
#pragma unroll
      for (int off = 1; off < 16; off <<= 1) {
        s += __shfl_xor(s, off, 32);
        w += __shfl_xor(w, off, 32);
        p += __shfl_xor(p, off, 32);
      }
      sk[h][k] = s; wk[h][k] = w; pk[h][k] = p;
    }
  if ((lane & 15) == 0) {
#pragma unroll
    for (int h = 0; h < 2; ++h)
#pragma unroll
      for (int k = 0; k < 8; ++k) {
        const int r = h * 16 + hi * 8 + k;
        st_s[wave][r] = sk[h][k];
        st_w[wave][r] = wk[h][k];
        st_p[wave][r] = pk[h][k];
      }
  }
  __syncthreads();

  // Merge the 8 wave-partials per row and accumulate the loss.
  if (tid < 32) {
    const int r = tid;
    float S = 0.0f, W = 0.0f, P = 0.0f;
#pragma unroll
    for (int w2 = 0; w2 < 8; ++w2) {
      S += st_s[w2][r];
      W += st_w[w2][r];
      P += st_p[w2][r];
    }
    const float mlpp = (P - W * (SHIFT + __logf(S + EPSF))) / (W + EPSF);
    atomicAdd(out, -mlpp * (1.0f / (float)N_B));   // T/baseT == 1
  }
}

// ---------------------------------------------------------------------------
extern "C" void kernel_launch(void* const* d_in, const int* in_sizes, int n_in,
                              void* d_out, int out_size, void* d_ws, size_t ws_size,
                              hipStream_t stream) {
  (void)in_sizes; (void)n_in; (void)out_size; (void)ws_size;
  const float* features = (const float*)d_in[0];
  const int*   labels   = (const int*)d_in[1];     // class ids 0..15
  const float* simw     = (const float*)d_in[2];   // [B][16]
  float*       out      = (float*)d_out;
  bf16_t*      fbf      = (bf16_t*)d_ws;           // [B][D] normalized bf16 (8 MB)

  wsc_zero_out<<<1, 32, 0, stream>>>(out);
  wsc_normalize<<<N_B, 256, 0, stream>>>(features, fbf);
  wsc_main<<<N_B / 32, 256, LDS_TOTAL, stream>>>(fbf, labels, simw, out);
}
